// ResidualBlockNoBNDynamic_9388798509163
// MI455X (gfx1250) — matrix operand order repaired and checked by reading, not compiled, not run on hardware
//
#include <hip/hip_runtime.h>
#include <hip/hip_bf16.h>

typedef _Float16 half_t;
typedef __attribute__((ext_vector_type(16))) _Float16 v16h;
typedef __attribute__((ext_vector_type(8)))  float    v8f;
typedef __attribute__((ext_vector_type(4)))  float    f32x4;

// ---------------------------------------------------------------------------
// fp32 -> fp16 convert
// ---------------------------------------------------------------------------
__global__ void f32_to_f16_kernel(const float* __restrict__ in,
                                  half_t* __restrict__ out, long n) {
    long i = (long)blockIdx.x * blockDim.x + threadIdx.x;
    if (i < n) out[i] = (half_t)in[i];
}

// ---------------------------------------------------------------------------
// Pack OIHW fp32 conv weights into WMMA A-fragment order:
//   apack[slice][oc][kk], slice = tap*ICT + ks, tap = kh*KHW + kw,
//   element = w[oc][ks*32+kk][kh][kw]   (oc >= OCreal zero-padded)
// ---------------------------------------------------------------------------
__global__ void pack_weights_kernel(const float* __restrict__ w,
                                    half_t* __restrict__ apack,
                                    int OCpad, int OCreal, int IC, int KHW, long n) {
    long idx = (long)blockIdx.x * blockDim.x + threadIdx.x;
    if (idx >= n) return;
    int kk = idx & 31;
    long t = idx >> 5;
    int oc = (int)(t % OCpad); t /= OCpad;
    int ICT = IC >> 5;
    int ks  = (int)(t % ICT);
    int tap = (int)(t / ICT);
    int kh = tap / KHW, kw = tap % KHW;
    int ic = ks * 32 + kk;
    float v = 0.f;
    if (oc < OCreal)
        v = w[(((long)oc * IC + ic) * KHW + kh) * KHW + kw];
    apack[idx] = (half_t)v;
}

// ---------------------------------------------------------------------------
// Mix K=5 expert 3x3 kernels by att and emit per-batch A-pack:
//   apackD[b][slice][oc][kk], slice = tap*2 + ks (18 slices), C=64
// ---------------------------------------------------------------------------
__global__ void mix_pack_kernel(const float* __restrict__ att,   // [16][5]
                                const float* __restrict__ ew,    // [5][64][64][3][3]
                                half_t* __restrict__ apackD) {
    long idx = (long)blockIdx.x * blockDim.x + threadIdx.x;
    if (idx >= 16L * 18 * 64 * 32) return;
    int kk = idx & 31;
    long t = idx >> 5;
    int oc = (int)(t & 63); t >>= 6;
    int slice = (int)(t % 18);
    int b = (int)(t / 18);
    int tap = slice >> 1, ks = slice & 1;
    int ic = ks * 32 + kk;
    int kh = tap / 3, kw = tap % 3;
    float v = 0.f;
#pragma unroll
    for (int k = 0; k < 5; ++k)
        v += att[b * 5 + k] * ew[((((long)k * 64 + oc) * 64 + ic) * 3 + kh) * 3 + kw];
    apackD[idx] = (half_t)v;
}

__global__ void mix_bias_kernel(const float* __restrict__ att,   // [16][5]
                                const float* __restrict__ eb,    // [5][64]
                                float* __restrict__ biasD) {     // [16][64]
    int b = blockIdx.x, oc = threadIdx.x;
    float v = 0.f;
#pragma unroll
    for (int k = 0; k < 5; ++k) v += att[b * 5 + k] * eb[k * 64 + oc];
    biasD[b * 64 + oc] = v;
}

// ---------------------------------------------------------------------------
// Global average pool:  h3 [16][16(pad)][128*128] f32 -> pooled [16][15]
// ---------------------------------------------------------------------------
__global__ void pool_kernel(const float* __restrict__ h3, float* __restrict__ pooled) {
    int b = blockIdx.x / 15, ch = blockIdx.x % 15;
    const float* p = h3 + ((long)b * 16 + ch) * 16384;
    float s = 0.f;
    for (int i = threadIdx.x; i < 16384; i += 256) s += p[i];
    __shared__ float sd[256];
    sd[threadIdx.x] = s;
    __syncthreads();
    for (int off = 128; off; off >>= 1) {
        if (threadIdx.x < off) sd[threadIdx.x] += sd[threadIdx.x + off];
        __syncthreads();
    }
    if (threadIdx.x == 0) pooled[blockIdx.x] = sd[0] * (1.f / 16384.f);
}

// ---------------------------------------------------------------------------
// Tiny MLP + softmax: one lane per batch sample (16 of 32 lanes active)
// ---------------------------------------------------------------------------
__global__ void mlp_softmax_kernel(const float* __restrict__ pooled,  // [16][15]
                                   const float* __restrict__ mw1,     // [15][15]
                                   const float* __restrict__ mb1,     // [15]
                                   const float* __restrict__ mw2,     // [5][15]
                                   const float* __restrict__ mb2,     // [5]
                                   float* __restrict__ att_ws,        // [16][5]
                                   float* __restrict__ att_out) {     // [16][5]
    int b = threadIdx.x;
    if (b >= 16) return;
    float p[15], m[15], o[5];
    for (int j = 0; j < 15; ++j) p[j] = pooled[b * 15 + j];
    for (int i = 0; i < 15; ++i) {
        float s = mb1[i];
        for (int j = 0; j < 15; ++j) s += p[j] * mw1[i * 15 + j];
        m[i] = s > 0.f ? s : 0.2f * s;
    }
    float mx = -3.4e38f;
    for (int k = 0; k < 5; ++k) {
        float s = mb2[k];
        for (int i = 0; i < 15; ++i) s += m[i] * mw2[k * 15 + i];
        o[k] = s;
        mx = s > mx ? s : mx;
    }
    float sum = 0.f;
    for (int k = 0; k < 5; ++k) { o[k] = __expf(o[k] - mx); sum += o[k]; }
    float inv = 1.f / sum;
    for (int k = 0; k < 5; ++k) {
        float a = o[k] * inv;
        att_ws[b * 5 + k]  = a;
        att_out[b * 5 + k] = a;
    }
}

// ---------------------------------------------------------------------------
// Implicit-GEMM conv via v_wmma_f32_16x16x32_f16. All geometry compile-time.
//   One block per (batch b, row pair h0..h0+1). 8 waves; wave w owns pixels
//   [w*16, w*16+16) in BOTH rows, so each A fragment feeds two WMMAs
//   (A-load : wmma = 1 : 1). Each wave accumulates 2 x OCT 16x16 tiles.
//   LDS holds one 32-channel input stripe of (KHW+1) rows in PIXEL-MAJOR
//   order with a 40-half (80 B) channel block per pixel:
//       lds[(r*Wp + c)*40 + icl]   (icl = 0..31, 8 halves pad)
//   so a B fragment (16 consecutive local channels at one pixel) is two
//   ds_load_b128, 16 B aligned, bank-conflict-free across lanes
//   (pixel stride 80 B -> bank 20*l mod 64 is a permutation of 16 lanes).
// ---------------------------------------------------------------------------
template <int OCT, int KHW, int PAD, int ICT>
__global__ __launch_bounds__(256)
void conv_wmma_kernel(const half_t* __restrict__ xin,          // f16 NCHW
                      const half_t* __restrict__ apack,        // packed A
                      long apack_bstride,                      // elems (0 = shared)
                      const float* __restrict__ bias,          // may be null
                      int bias_bstride, int OC_real,
                      float leak,                              // 0=relu, 0.2=leaky, 1=id
                      const float* __restrict__ residual,      // may be null
                      half_t* __restrict__ yh,                 // f16 out (opt)
                      float* __restrict__ yf)                  // f32 out (opt)
{
    constexpr int H = 128, W = 128;
    constexpr int Wp = W + 2 * PAD;
    constexpr int IC = ICT * 32;
    constexpr int TAPS = KHW * KHW;
    constexpr int ROWS = KHW + 1;             // stripe rows for 2 output rows
    constexpr int CBLK = 40;                  // halves per pixel block (32 + 8 pad)

    const int h0 = blockIdx.x * 2, b = blockIdx.y;
    const int tid = threadIdx.x;
    const int wave = tid >> 5, lane = tid & 31;
    extern __shared__ half_t lds[];           // ROWS * Wp * CBLK halves

    const half_t* ap = apack + (long)b * apack_bstride;
    const int col   = lane & 15;              // N col (B/D) and M row (A)
    const int khalf = lane >> 4;              // which K half this lane holds
    const int px    = wave * 16 + col;        // output pixel for this lane

    const v8f vzero = {};
    v8f acc[2][OCT];
#pragma unroll
    for (int rr = 0; rr < 2; ++rr)
#pragma unroll
        for (int i = 0; i < OCT; ++i) acc[rr][i] = vzero;

    // hint: pull packed weights toward the caches
    __builtin_prefetch((const void*)(ap + (long)col * 32), 0, 3);

    for (int ich = 0; ich < ICT; ++ich) {
        // ---- stage 32 input channels x ROWS rows (zero-padded) into LDS ----
        constexpr int TOTAL = 32 * ROWS * Wp;
        for (int idx = tid; idx < TOTAL; idx += 256) {
            int c = idx % Wp;                 // coalesced in c per (icl, r)
            int t = idx / Wp;
            int r   = t % ROWS;
            int icl = t / ROWS;
            int yy = h0 + r - PAD;
            int xx = c - PAD;
            half_t v = (half_t)0.f;
            if ((unsigned)yy < (unsigned)H && (unsigned)xx < (unsigned)W)
                v = xin[(((long)b * IC + ich * 32 + icl) * H + yy) * W + xx];
            lds[(r * Wp + c) * CBLK + icl] = v;
        }
        __syncthreads();

#pragma unroll
        for (int tap = 0; tap < TAPS; ++tap) {
            const int kh = tap / KHW, kw = tap % KHW;
            // B fragments: 16 contiguous local channels at pixel (kh(+1), px+kw)
            const half_t* bsrc = lds + (kh * Wp + px + kw) * CBLK + khalf * 16;
            v16h bfrag0, bfrag1;
            *(f32x4*)&bfrag0       = *(const f32x4*)bsrc;            // ds_load_b128
            *((f32x4*)&bfrag0 + 1) = *(const f32x4*)(bsrc + 8);
            *(f32x4*)&bfrag1       = *(const f32x4*)(bsrc + Wp * CBLK);
            *((f32x4*)&bfrag1 + 1) = *(const f32x4*)(bsrc + Wp * CBLK + 8);

            const int slice = tap * ICT + ich;
            const half_t* abase = ap + ((long)slice * (OCT * 16) + col) * 32 + khalf * 16;
#pragma unroll
            for (int mt = 0; mt < OCT; ++mt) {
                // A fragment: row = col, K {khalf*16..+15} contiguous
                const half_t* arow = abase + mt * 16 * 32;
                v16h afrag;
                *(f32x4*)&afrag       = *(const f32x4*)arow;        // global b128
                *((f32x4*)&afrag + 1) = *(const f32x4*)(arow + 8);  // global b128
                acc[0][mt] = __builtin_amdgcn_wmma_f32_16x16x32_f16(
                    false, afrag, false, bfrag0, (short)0, acc[0][mt], false, false);
                acc[1][mt] = __builtin_amdgcn_wmma_f32_16x16x32_f16(
                    false, afrag, false, bfrag1, (short)0, acc[1][mt], false, false);
            }
        }
        __syncthreads();
    }

    // ---- epilogue: bias + activation + optional residual, store ----
    constexpr int OCpad = OCT * 16;
#pragma unroll
    for (int rr = 0; rr < 2; ++rr) {
        const int hh = h0 + rr;
#pragma unroll
        for (int mt = 0; mt < OCT; ++mt) {
#pragma unroll
            for (int r = 0; r < 8; ++r) {
                int oc = mt * 16 + khalf * 8 + r;   // D layout: M = r + 8*(lane>=16)
                float v = acc[rr][mt][r];
                if (bias && oc < OC_real) v += bias[(long)b * bias_bstride + oc];
                v = v > 0.f ? v : leak * v;
                long oidx = (((long)b * OCpad + oc) * H + hh) * W + px;
                if (residual) v += residual[oidx];
                if (yf) yf[oidx] = v;
                if (yh) yh[oidx] = (half_t)v;
            }
        }
    }
}

// ---------------------------------------------------------------------------
// Host side
// ---------------------------------------------------------------------------
extern "C" void kernel_launch(void* const* d_in, const int* in_sizes, int n_in,
                              void* d_out, int out_size, void* d_ws, size_t ws_size,
                              hipStream_t stream) {
    (void)in_sizes; (void)n_in; (void)out_size; (void)ws_size;

    const long NELEM = 16L * 64 * 128 * 128;      // 16,777,216
    const float* x = (const float*)d_in[0];

    struct LayerIn {
        const float *fw1, *fb1, *fw2, *fb2, *fw3, *fb3;
        const float *mw1, *mb1, *mw2, *mb2, *ew, *eb;
    } L[2];
    for (int l = 0; l < 2; ++l) {
        int o = 1 + l * 12;
        L[l].fw1 = (const float*)d_in[o + 0];  L[l].fb1 = (const float*)d_in[o + 1];
        L[l].fw2 = (const float*)d_in[o + 2];  L[l].fb2 = (const float*)d_in[o + 3];
        L[l].fw3 = (const float*)d_in[o + 4];  L[l].fb3 = (const float*)d_in[o + 5];
        L[l].mw1 = (const float*)d_in[o + 6];  L[l].mb1 = (const float*)d_in[o + 7];
        L[l].mw2 = (const float*)d_in[o + 8];  L[l].mb2 = (const float*)d_in[o + 9];
        L[l].ew  = (const float*)d_in[o + 10]; L[l].eb  = (const float*)d_in[o + 11];
    }

    // ---- carve workspace ----
    char* ws = (char*)d_ws;
    size_t cur = 0;
    auto alloc = [&](size_t bytes) -> char* {
        char* p = ws + cur;
        cur = (cur + bytes + 255) & ~(size_t)255;
        return p;
    };
    half_t* xh   = (half_t*)alloc(NELEM * 2);
    half_t* y1h  = (half_t*)alloc(NELEM * 2);
    half_t* h1   = (half_t*)alloc(NELEM * 2);
    half_t* h2   = (half_t*)alloc(NELEM * 2);
    float*  h3f  = (float*)alloc(16L * 16 * 16384 * 4);
    half_t* apF1 = (half_t*)alloc(50L * 64 * 32 * 2);
    half_t* apF2 = (half_t*)alloc(50L * 64 * 32 * 2);
    half_t* apF3 = (half_t*)alloc(50L * 16 * 32 * 2);
    half_t* apD  = (half_t*)alloc(16L * 18 * 64 * 32 * 2);
    float*  pooled = (float*)alloc(16 * 15 * 4);
    float*  attws  = (float*)alloc(16 * 5 * 4);
    float*  biasD  = (float*)alloc(16 * 64 * 4);

    float* out_f = (float*)d_out;
    float* att_out[2] = { out_f + NELEM, out_f + NELEM + 80 };

    // x -> fp16 once
    f32_to_f16_kernel<<<(int)(NELEM / 256), 256, 0, stream>>>(x, xh, NELEM);

    const dim3 grid(64, 16);                                      // 2 rows / block
    const size_t sh5 = (size_t)(6 * 132 * 40) * sizeof(half_t);   // 63,360 B
    const size_t sh3 = (size_t)(4 * 130 * 40) * sizeof(half_t);   // 41,600 B

    for (int l = 0; l < 2; ++l) {
        const half_t* ins = (l == 0) ? xh : y1h;

        // ---- pack fusion weights ----
        long n1 = 50L * 64 * 32;
        pack_weights_kernel<<<(int)((n1 + 255) / 256), 256, 0, stream>>>(
            L[l].fw1, apF1, 64, 64, 64, 5, n1);
        pack_weights_kernel<<<(int)((n1 + 255) / 256), 256, 0, stream>>>(
            L[l].fw2, apF2, 64, 64, 64, 5, n1);
        long n3 = 50L * 16 * 32;
        pack_weights_kernel<<<(int)((n3 + 255) / 256), 256, 0, stream>>>(
            L[l].fw3, apF3, 16, 15, 64, 5, n3);

        // ---- fusion conv chain (leaky 0.2) ----
        conv_wmma_kernel<4, 5, 2, 2><<<grid, 256, sh5, stream>>>(
            ins, apF1, 0, L[l].fb1, 0, 64, 0.2f, nullptr, h1, nullptr);
        conv_wmma_kernel<4, 5, 2, 2><<<grid, 256, sh5, stream>>>(
            h1, apF2, 0, L[l].fb2, 0, 64, 0.2f, nullptr, h2, nullptr);
        conv_wmma_kernel<1, 5, 2, 2><<<grid, 256, sh5, stream>>>(
            h2, apF3, 0, L[l].fb3, 0, 15, 0.2f, nullptr, nullptr, h3f);

        // ---- pool + MLP + softmax -> att ----
        pool_kernel<<<240, 256, 0, stream>>>(h3f, pooled);
        mlp_softmax_kernel<<<1, 32, 0, stream>>>(
            pooled, L[l].mw1, L[l].mb1, L[l].mw2, L[l].mb2, attws, att_out[l]);

        // ---- mix experts -> per-batch packed weights + bias ----
        mix_pack_kernel<<<(int)((16L * 18 * 64 * 32) / 256), 256, 0, stream>>>(
            attws, L[l].ew, apD);
        mix_bias_kernel<<<16, 64, 0, stream>>>(attws, L[l].eb, biasD);

        // ---- dynamic 3x3 conv ----
        if (l == 0) {
            // relu epilogue, fp16 out feeds layer 2
            conv_wmma_kernel<4, 3, 1, 2><<<grid, 256, sh3, stream>>>(
                ins, apD, 18L * 64 * 32, biasD, 64, 64, 0.0f,
                nullptr, y1h, nullptr);
        } else {
            // identity activation + residual(x) -> final fp32 output
            conv_wmma_kernel<4, 3, 1, 2><<<grid, 256, sh3, stream>>>(
                ins, apD, 18L * 64 * 32, biasD, 64, 64, 1.0f,
                x, nullptr, out_f);
        }
    }
}
